// SimplifiedReportGenerationLoss_51866025067157
// MI455X (gfx1250) — 8-bit matrix operand order repaired and checked by reading, compiled, not run on hardware
//
#include <hip/hip_runtime.h>
#include <stdint.h>

typedef int v8i __attribute__((ext_vector_type(8)));

__global__ void zero_ws_kernel(uint32_t* w, int nwords) {
    int i = blockIdx.x * blockDim.x + threadIdx.x;
    int stride = gridDim.x * blockDim.x;
    for (; i < nwords; i += stride) w[i] = 0u;
}

__global__ void hist_membership_kernel(const int* __restrict__ tok,
                                       uint32_t* __restrict__ hist,
                                       uint8_t* __restrict__ M,
                                       int BS, int S, int V) {
    int i = blockIdx.x * blockDim.x + threadIdx.x;
    if (i >= BS) return;
    int t = tok[i];
    if ((unsigned)t < (unsigned)V) {
        atomicAdd(&hist[t], 1u);
        int b = i / S;
        M[(size_t)b * V + t] = (uint8_t)1;
    }
}

__global__ void entropy_unigram_kernel(const uint32_t* __restrict__ hist, int V,
                                       float invDenom, float* entAcc, int* uniq1) {
    int i = blockIdx.x * blockDim.x + threadIdx.x;
    if (i >= V) return;
    uint32_t c = hist[i];
    if (c > 0u) {
        atomicAdd(uniq1, 1);
        float p = (float)c * invDenom;
        atomicAdd(entAcc, p * logf(p + 1e-8f));
    }
}

__global__ void bigram_codes_kernel(const int* __restrict__ tok,
                                    uint32_t* __restrict__ big, int B, int S, int V) {
    int n = B * (S - 1);
    int i = blockIdx.x * blockDim.x + threadIdx.x;
    if (i >= n) return;
    int b = i / (S - 1), s = i % (S - 1);
    uint32_t t0 = (uint32_t)tok[b * S + s];
    uint32_t t1 = (uint32_t)tok[b * S + s + 1];
    big[i] = t0 * (uint32_t)V + t1;
}

__global__ void trigram_codes_kernel(const int* __restrict__ tok,
                                     const uint32_t* __restrict__ big,
                                     uint64_t* __restrict__ tri, int B, int S, int V) {
    int n = B * (S - 2);
    int i = blockIdx.x * blockDim.x + threadIdx.x;
    if (i >= n) return;
    int b = i / (S - 2), s = i % (S - 2);
    uint64_t bg = (uint64_t)big[b * (S - 1) + s];
    tri[i] = bg * (uint64_t)(uint32_t)V + (uint64_t)(uint32_t)tok[b * S + s + 2];
}

#define NGMAX 8192
__global__ void bigram_uniq_lds_kernel(const uint32_t* __restrict__ big, int n, int* uniq) {
    __shared__ uint32_t sc[NGMAX];
    for (int j = threadIdx.x; j < n; j += blockDim.x) sc[j] = big[j];
    __syncthreads();
    int i = blockIdx.x * blockDim.x + threadIdx.x;
    if (i >= n) return;
    uint32_t c = sc[i];
    for (int j = 0; j < i; ++j) if (sc[j] == c) return;
    atomicAdd(uniq, 1);
}

__global__ void uniq_u32_global_kernel(const uint32_t* __restrict__ codes, int n, int* uniq) {
    int i = blockIdx.x * blockDim.x + threadIdx.x;
    if (i >= n) return;
    uint32_t c = codes[i];
    for (int j = 0; j < i; ++j) if (codes[j] == c) return;
    atomicAdd(uniq, 1);
}

__global__ void uniq_u64_global_kernel(const uint64_t* __restrict__ codes, int n, int* uniq) {
    int i = blockIdx.x * blockDim.x + threadIdx.x;
    if (i >= n) return;
    uint64_t c = codes[i];
    for (int j = 0; j < i; ++j) if (codes[j] == c) return;
    atomicAdd(uniq, 1);
}

__global__ void trigram_rowuniq_kernel(const uint64_t* __restrict__ tri, int B, int L,
                                       int* __restrict__ uniqRows) {
    int i = blockIdx.x * blockDim.x + threadIdx.x;
    if (i >= B * L) return;
    int b = i / L, s = i % L;
    const uint64_t* row = tri + (size_t)b * L;
    uint64_t c = row[s];
    for (int j = 0; j < s; ++j) if (row[j] == c) return;
    atomicAdd(&uniqRows[b], 1);
}

__global__ void selfbleu_wmma_kernel(const uint8_t* __restrict__ M, int V, float* sbOut) {
    __shared__ int inter[16][16];
    const int lane = threadIdx.x;
    const int r    = lane & 15;
    const int hi   = lane >> 4;
    const uint32_t* rowp = reinterpret_cast<const uint32_t*>(M + (size_t)r * V);

    const int aoff[8] = {0, 4, 16, 20, 32, 36, 48, 52};
    const int boff[8] = {0, 4, 16, 20, 32, 36, 48, 52};
    const int abase = hi ? 8 : 0;
    const int bbase = hi ? 8 : 0;

    v8i acc = {0, 0, 0, 0, 0, 0, 0, 0};
    const int kmax = V - (V & 63);
    for (int k0 = 0; k0 < kmax; k0 += 64) {
        v8i av, bv;
#pragma unroll
        for (int v = 0; v < 8; ++v) {
            av[v] = (int)rowp[(k0 + abase + aoff[v]) >> 2];
            bv[v] = (int)rowp[(k0 + bbase + boff[v]) >> 2];
        }
        acc = __builtin_amdgcn_wmma_i32_16x16x64_iu8(false, av, false, bv, acc,
                                                     false, false);
    }
#pragma unroll
    for (int v = 0; v < 8; ++v) inter[v + 8 * hi][r] = acc[v];
    __syncthreads();

    if (lane == 0) {
        float s = 0.f;
        for (int i = 0; i < 10; ++i) {
            int d = inter[i][i];
            float ref = (float)(d > 1 ? d : 1);
            for (int j = 0; j < 16; ++j)
                if (j != i) s += (float)inter[i][j] / ref;
        }
        *sbOut = s / 150.f;
    }
}

__global__ void conf_kernel(const float* __restrict__ logits, int V, float* confAcc) {
    __shared__ float sm[256];
    __shared__ float ss[256];
    const int t = threadIdx.x;
    const float* row = logits + (size_t)blockIdx.x * V;
    const float4* row4 = reinterpret_cast<const float4*>(row);
    const int n4 = V >> 2;

    float m = -INFINITY, s = 0.f;
    for (int i = t; i < n4; i += 256) {
        __builtin_prefetch(&row4[i + 1024], 0, 1);
        float4 x4 = row4[i];
        float xs[4] = {x4.x, x4.y, x4.z, x4.w};
#pragma unroll
        for (int v = 0; v < 4; ++v) {
            float x = xs[v];
            if (x > m) { s = s * __expf(m - x) + 1.0f; m = x; }
            else       { s += __expf(x - m); }
        }
    }
    if (t == 0) {
        for (int k = n4 << 2; k < V; ++k) {
            float x = row[k];
            if (x > m) { s = s * __expf(m - x) + 1.0f; m = x; }
            else       { s += __expf(x - m); }
        }
    }
    sm[t] = m; ss[t] = s;
    __syncthreads();
    for (int off = 128; off > 0; off >>= 1) {
        if (t < off) {
            float m1 = sm[t], s1 = ss[t];
            float m2 = sm[t + off], s2 = ss[t + off];
            float nm = fmaxf(m1, m2);
            float w1 = (m1 == -INFINITY) ? 0.f : __expf(m1 - nm);
            float w2 = (m2 == -INFINITY) ? 0.f : __expf(m2 - nm);
            sm[t] = nm; ss[t] = s1 * w1 + s2 * w2;
        }
        __syncthreads();
    }
    if (t == 0) atomicAdd(confAcc, 1.0f / ss[0]);
}

__global__ void finalize_kernel(const float* __restrict__ accF,
                                const int* __restrict__ accI,
                                int B, int S, int V, float* out) {
    if (threadIdx.x != 0 || blockIdx.x != 0) return;
    float d1 = (float)accI[0] / (float)(B * S);
    float d2 = (float)accI[1] / (float)(B * (S - 1));
    float d3 = (float)accI[2] / (float)(B * (S - 2));
    float ngram = ((1.f - d1) + (1.f - d2) + (1.f - d3)) / 3.f;

    float entropy = -accF[0];
    float tokent = 1.f - entropy / logf((float)V);

    float sb = accF[2];

    float rep = 0.f;
    for (int b = 0; b < B; ++b)
        rep += 1.f - (float)accI[3 + b] / (float)(S - 2);
    rep /= (float)B;

    float avg = accF[1] / (float)(B * S);
    float oc = fmaxf(avg - 0.85f, 0.f) * 2.f;

    out[0] = 0.25f * ngram + 0.2f * tokent + 0.2f * sb + 0.2f * rep + 0.15f * oc;
}

extern "C" void kernel_launch(void* const* d_in, const int* in_sizes, int n_in,
                              void* d_out, int out_size, void* d_ws, size_t ws_size,
                              hipStream_t stream) {
    const int*   tok    = (const int*)d_in[0];
    const float* logits = (const float*)d_in[1];

    const int BS = in_sizes[0];
    const int V  = (int)((long long)in_sizes[1] / (long long)BS);
    const int B  = 16;
    const int S  = BS / B;

    char* ws = (char*)d_ws;
    float* accF = (float*)ws;
    int*   accI = (int*)(ws + 32);
    size_t histOff = 128;
    uint32_t* hist = (uint32_t*)(ws + histOff);
    size_t bigOff = histOff + (size_t)V * 4;
    uint32_t* big = (uint32_t*)(ws + bigOff);
    size_t triOff = (bigOff + (size_t)B * (S - 1) * 4 + 7) & ~(size_t)7;
    uint64_t* tri = (uint64_t*)(ws + triOff);
    size_t mOff = triOff + (size_t)B * (S - 2) * 8;
    uint8_t* M = (uint8_t*)(ws + mOff);

    const int n2 = B * (S - 1);
    const int n3 = B * (S - 2);

    int zw1 = (int)(bigOff / 4);
    zero_ws_kernel<<<(zw1 + 255) / 256, 256, 0, stream>>>((uint32_t*)ws, zw1);
    int zw2 = (int)(((size_t)B * V + 3) / 4);
    zero_ws_kernel<<<(zw2 + 255) / 256, 256, 0, stream>>>((uint32_t*)(ws + mOff), zw2);

    hist_membership_kernel<<<(BS + 255) / 256, 256, 0, stream>>>(tok, hist, M, BS, S, V);

    float invDenom = (float)(1.0 / ((double)BS + 1e-8));
    entropy_unigram_kernel<<<(V + 255) / 256, 256, 0, stream>>>(hist, V, invDenom,
                                                                &accF[0], &accI[0]);

    bigram_codes_kernel<<<(n2 + 255) / 256, 256, 0, stream>>>(tok, big, B, S, V);
    trigram_codes_kernel<<<(n3 + 255) / 256, 256, 0, stream>>>(tok, big, tri, B, S, V);

    if (n2 <= NGMAX)
        bigram_uniq_lds_kernel<<<(n2 + 255) / 256, 256, 0, stream>>>(big, n2, &accI[1]);
    else
        uniq_u32_global_kernel<<<(n2 + 255) / 256, 256, 0, stream>>>(big, n2, &accI[1]);

    uniq_u64_global_kernel<<<(n3 + 255) / 256, 256, 0, stream>>>(tri, n3, &accI[2]);
    trigram_rowuniq_kernel<<<(n3 + 255) / 256, 256, 0, stream>>>(tri, B, S - 2, &accI[3]);

    selfbleu_wmma_kernel<<<1, 32, 0, stream>>>(M, V, &accF[2]);

    conf_kernel<<<BS, 256, 0, stream>>>(logits, V, &accF[1]);

    finalize_kernel<<<1, 1, 0, stream>>>(accF, accI, B, S, V, (float*)d_out);
}
